// SelfAttention_6408091206042
// MI455X (gfx1250) — matrix operand order repaired and checked by reading, not compiled, not run on hardware
//
#include <hip/hip_runtime.h>
#include <hip/hip_bf16.h>

// ---- CDNA5 WMMA types -------------------------------------------------------
typedef __attribute__((ext_vector_type(16))) _Float16 v16h;
typedef __attribute__((ext_vector_type(8)))  _Float16 v8h;
typedef __attribute__((ext_vector_type(8)))  float    v8f;
typedef __attribute__((ext_vector_type(4)))  int      v4i;

union Frag16 {
    v16h v;
    v8h  h[2];
    _Float16 e[16];
};

#define WMMA_F16(A, B, C) \
    __builtin_amdgcn_wmma_f32_16x16x32_f16(false, (A), false, (B), (short)0, (C), false, false)

// Problem constants
#define BATCH 8
#define CDIM  512
#define NSEQ  1024       // 32*32
#define NHEAD 8
#define HDIM  64

// ---- CDNA5 async global->LDS copy (guarded; sync fallback keeps compile OK) --
__device__ __forceinline__ void async_copy_b128(const _Float16* g, _Float16* l)
{
#if __has_builtin(__builtin_amdgcn_global_load_async_to_lds_b128)
    typedef __attribute__((address_space(1))) v4i gv4_t;   // global 16B vector
    typedef __attribute__((address_space(3))) v4i lv4_t;   // LDS 16B vector
    __builtin_amdgcn_global_load_async_to_lds_b128((gv4_t*)g, (lv4_t*)l, 0, 0);
#else
    *(v8h*)l = *(const v8h*)g;
#endif
}

__device__ __forceinline__ void wait_async_lds()
{
#if __has_builtin(__builtin_amdgcn_s_wait_asynccnt)
    __builtin_amdgcn_s_wait_asynccnt(0);
#else
    asm volatile("s_wait_asynccnt 0" ::: "memory");
#endif
}

// ---------------------------------------------------------------------------
// Kernel 1: x [B,C,N] f32 -> xt [B,N,C] f16  (tiled transpose + convert)
// ---------------------------------------------------------------------------
__global__ __launch_bounds__(256) void k_transpose_convert(
    const float* __restrict__ x, _Float16* __restrict__ xt)
{
    __shared__ float tile[32][33];
    const int b  = blockIdx.z;
    const int n0 = blockIdx.x * 32;
    const int c0 = blockIdx.y * 32;
    const int tx = threadIdx.x;
    const int ty = threadIdx.y;

    #pragma unroll
    for (int i = 0; i < 32; i += 8)
        tile[ty + i][tx] = x[((size_t)(b * CDIM + c0 + ty + i)) * NSEQ + n0 + tx];

    __syncthreads();

    #pragma unroll
    for (int i = 0; i < 32; i += 8)
        xt[((size_t)(b * NSEQ + n0 + ty + i)) * CDIM + c0 + tx] =
            (_Float16)tile[tx][ty + i];
}

// ---------------------------------------------------------------------------
// Kernel 2: elementwise f32 -> f16 (weights)
// ---------------------------------------------------------------------------
__global__ __launch_bounds__(256) void k_convert_f16(
    const float* __restrict__ w, _Float16* __restrict__ wh, int n)
{
    int i = blockIdx.x * 256 + threadIdx.x;
    if (i < n) wh[i] = (_Float16)w[i];
}

// ---------------------------------------------------------------------------
// Kernel 3: Q/K projection.  Out[n][o] = sum_c xt[b][n][c]*W[o][c] + bias[o]
// One wave -> 32 n x 64 o (2x4 WMMA tiles): each B fragment feeds 2 WMMAs.
// ---------------------------------------------------------------------------
__global__ __launch_bounds__(128) void k_proj_qk(
    const _Float16* __restrict__ xt, const _Float16* __restrict__ Wh,
    const float* __restrict__ bias, _Float16* __restrict__ out, float scale)
{
    const int lane = threadIdx.x & 31;
    const int wv   = threadIdx.x >> 5;
    const int waveId = blockIdx.x * 4 + wv;          // 2048 waves
    const int ogrp  = waveId & 7;                    // 8 * 64 = 512 o
    const int ntile = (waveId >> 3) & 31;            // 32 * 32 = 1024 n
    const int b     = waveId >> 8;                   // 8
    const int lo = lane & 15, hi = lane >> 4;
    const int n0 = ntile * 32, o0 = ogrp * 64;

    v8f acc0[4] = {}, acc1[4] = {};
    const _Float16* arow0 = xt + ((size_t)(b * NSEQ + n0 + lo)) * CDIM;
    const _Float16* arow1 = arow0 + (size_t)16 * CDIM;

    for (int k0 = 0; k0 < CDIM; k0 += 32) {
        Frag16 a0, a1;
        a0.h[0] = *(const v8h*)(arow0 + k0 + hi * 8);
        a0.h[1] = *(const v8h*)(arow0 + k0 + 16 + hi * 8);
        a1.h[0] = *(const v8h*)(arow1 + k0 + hi * 8);
        a1.h[1] = *(const v8h*)(arow1 + k0 + 16 + hi * 8);
        #pragma unroll
        for (int j = 0; j < 4; ++j) {
            Frag16 bf;
            const _Float16* brow = Wh + ((size_t)(o0 + j * 16 + lo)) * CDIM + k0 + hi * 8;
            bf.h[0] = *(const v8h*)(brow);
            bf.h[1] = *(const v8h*)(brow + 16);
            acc0[j] = WMMA_F16(a0.v, bf.v, acc0[j]);
            acc1[j] = WMMA_F16(a1.v, bf.v, acc1[j]);
        }
    }
    #pragma unroll
    for (int j = 0; j < 4; ++j) {
        const int o  = o0 + j * 16 + lo;
        const float bv = bias[o];
        const int h  = o >> 6, dd = o & 63;
        #pragma unroll
        for (int r = 0; r < 8; ++r) {
            const int n = n0 + r + 8 * hi;
            out[(((size_t)(b * NHEAD + h)) * NSEQ + n) * HDIM + dd] =
                (_Float16)((acc0[j][r] + bv) * scale);
            out[(((size_t)(b * NHEAD + h)) * NSEQ + n + 16) * HDIM + dd] =
                (_Float16)((acc1[j][r] + bv) * scale);
        }
    }
}

// ---------------------------------------------------------------------------
// Kernel 4: V projection.  Out[o][n] = sum_c W[o][c]*xt[b][n][c] + bias[o]
// One wave -> 32 o x 64 n.  Stored [bh,d,N].
// ---------------------------------------------------------------------------
__global__ __launch_bounds__(128) void k_proj_v(
    const _Float16* __restrict__ xt, const _Float16* __restrict__ Wh,
    const float* __restrict__ bias, _Float16* __restrict__ out)
{
    const int lane = threadIdx.x & 31;
    const int wv   = threadIdx.x >> 5;
    const int waveId = blockIdx.x * 4 + wv;          // 2048 waves
    const int ngrp  = waveId & 15;                   // 16 * 64 = 1024 n
    const int otile = (waveId >> 4) & 15;            // 16 * 32 = 512 o
    const int b     = waveId >> 8;
    const int lo = lane & 15, hi = lane >> 4;
    const int o0 = otile * 32, n0 = ngrp * 64;

    v8f acc0[4] = {}, acc1[4] = {};
    const _Float16* arow0 = Wh + ((size_t)(o0 + lo)) * CDIM;
    const _Float16* arow1 = arow0 + (size_t)16 * CDIM;

    for (int k0 = 0; k0 < CDIM; k0 += 32) {
        Frag16 a0, a1;
        a0.h[0] = *(const v8h*)(arow0 + k0 + hi * 8);
        a0.h[1] = *(const v8h*)(arow0 + k0 + 16 + hi * 8);
        a1.h[0] = *(const v8h*)(arow1 + k0 + hi * 8);
        a1.h[1] = *(const v8h*)(arow1 + k0 + 16 + hi * 8);
        #pragma unroll
        for (int j = 0; j < 4; ++j) {
            Frag16 bf;
            const _Float16* brow = xt + ((size_t)(b * NSEQ + n0 + j * 16 + lo)) * CDIM + k0 + hi * 8;
            bf.h[0] = *(const v8h*)(brow);
            bf.h[1] = *(const v8h*)(brow + 16);
            acc0[j] = WMMA_F16(a0.v, bf.v, acc0[j]);
            acc1[j] = WMMA_F16(a1.v, bf.v, acc1[j]);
        }
    }
    #pragma unroll
    for (int r = 0; r < 8; ++r) {
        #pragma unroll
        for (int rb = 0; rb < 2; ++rb) {
            const int o  = o0 + rb * 16 + r + 8 * hi;
            const float bv = bias[o];
            const int h  = o >> 6, dd = o & 63;
            #pragma unroll
            for (int j = 0; j < 4; ++j) {
                const int n = n0 + j * 16 + lo;
                const float v = (rb ? acc1[j][r] : acc0[j][r]) + bv;
                out[(((size_t)(b * NHEAD + h)) * HDIM + dd) * NSEQ + n] = (_Float16)v;
            }
        }
    }
}

// ---------------------------------------------------------------------------
// Kernel 5: flash attention.  Block = 4 waves, all on the SAME (b,h), with 4
// adjacent 16-row query tiles.  The 32-key K/V block is staged into LDS once
// per block with async global->LDS b128 copies (double-buffered), then each
// wave gathers WMMA fragments from LDS via ds_load_b128.
// ---------------------------------------------------------------------------
__global__ __launch_bounds__(128) void k_attention(
    const _Float16* __restrict__ Q, const _Float16* __restrict__ K,
    const _Float16* __restrict__ V, _Float16* __restrict__ att)
{
    __shared__ _Float16 ktile[2][32 * 64];  // [key][d]
    __shared__ _Float16 vtile[2][64 * 32];  // [d][key]
    __shared__ _Float16 pbuf[4][16 * 32];

    const int tid  = threadIdx.x;
    const int lane = tid & 31;
    const int wv   = tid >> 5;
    const int bh   = blockIdx.x >> 4;                // 64 (b,h)
    const int qgrp = blockIdx.x & 15;                // 16 query groups of 64
    const int lo = lane & 15, hi = lane >> 4;
    const int n0 = qgrp * 64 + wv * 16;

    const _Float16* qbase = Q + ((size_t)bh) * NSEQ * HDIM;
    const _Float16* kbase = K + ((size_t)bh) * NSEQ * HDIM;
    const _Float16* vbase = V + ((size_t)bh) * HDIM * NSEQ;
    _Float16* pp = &pbuf[wv][0];

    // Q fragments (A-layout): row = n0+lo, two 32-wide K chunks over d
    Frag16 aq[2];
    const _Float16* qrow = qbase + ((size_t)(n0 + lo)) * HDIM;
    #pragma unroll
    for (int t = 0; t < 2; ++t) {
        aq[t].h[0] = *(const v8h*)(qrow + t * 32 + hi * 8);
        aq[t].h[1] = *(const v8h*)(qrow + t * 32 + 16 + hi * 8);
    }

    v8f o[4] = {};
    float m[8], l[8];
    #pragma unroll
    for (int r = 0; r < 8; ++r) { m[r] = -3.0e38f; l[r] = 0.0f; }

    // cooperative staging of one 32-key K/V block (8 KB) by 128 threads
    const int krow = tid >> 2, kcol = (tid & 3) * 16;   // K: 32 x 64
    const int vrow = tid >> 1, vcol = (tid & 1) * 16;   // V: 64 x 32
    auto stage = [&](int buf, int kb) {
        const _Float16* gk = kbase + ((size_t)(kb + krow)) * HDIM + kcol;
        _Float16* lk = &ktile[buf][krow * 64 + kcol];
        async_copy_b128(gk, lk);
        async_copy_b128(gk + 8, lk + 8);
        const _Float16* gv = vbase + ((size_t)vrow) * NSEQ + kb + vcol;
        _Float16* lv = &vtile[buf][vrow * 32 + vcol];
        async_copy_b128(gv, lv);
        async_copy_b128(gv + 8, lv + 8);
    };

    stage(0, 0);
    int ib = 0;
    for (int kb = 0; kb < NSEQ; kb += 32, ib ^= 1) {
        wait_async_lds();
        __syncthreads();                      // tile ib ready for everyone
        if (kb + 32 < NSEQ) stage(ib ^ 1, kb + 32);

        const _Float16* kt = &ktile[ib][0];
        const _Float16* vt = &vtile[ib][0];

        // ---- S = Q * K^T for 32 keys (two 16-col tiles) from LDS ----
        v8f s[2];
        #pragma unroll
        for (int t = 0; t < 2; ++t) {
            const _Float16* krw = kt + (size_t)(t * 16 + lo) * 64;
            Frag16 bk0, bk1;
            bk0.h[0] = *(const v8h*)(krw + hi * 8);
            bk0.h[1] = *(const v8h*)(krw + 16 + hi * 8);
            bk1.h[0] = *(const v8h*)(krw + 32 + hi * 8);
            bk1.h[1] = *(const v8h*)(krw + 32 + 16 + hi * 8);
            v8f z = {};
            z    = WMMA_F16(aq[0].v, bk0.v, z);
            s[t] = WMMA_F16(aq[1].v, bk1.v, z);
        }
        // ---- online softmax update (per-lane rows r + 8*hi) ----
        #pragma unroll
        for (int r = 0; r < 8; ++r) {
            float mx = fmaxf(s[0][r], s[1][r]);
            mx = fmaxf(mx, __shfl_xor(mx, 1, 32));
            mx = fmaxf(mx, __shfl_xor(mx, 2, 32));
            mx = fmaxf(mx, __shfl_xor(mx, 4, 32));
            mx = fmaxf(mx, __shfl_xor(mx, 8, 32));
            const float nm = fmaxf(m[r], mx);
            const float sc = __expf(m[r] - nm);
            const float p0 = __expf(s[0][r] - nm);
            const float p1 = __expf(s[1][r] - nm);
            float rs = p0 + p1;
            rs += __shfl_xor(rs, 1, 32);
            rs += __shfl_xor(rs, 2, 32);
            rs += __shfl_xor(rs, 4, 32);
            rs += __shfl_xor(rs, 8, 32);
            l[r] = l[r] * sc + rs;
            m[r] = nm;
            #pragma unroll
            for (int dt = 0; dt < 4; ++dt) o[dt][r] *= sc;
            const int row = r + 8 * hi;
            pp[row * 32 + lo]      = (_Float16)p0;
            pp[row * 32 + 16 + lo] = (_Float16)p1;
        }
        asm volatile("s_wait_dscnt 0" ::: "memory");
        // re-gather P in A-fragment order (wave-local LDS tile)
        Frag16 pa;
        pa.h[0] = *(const v8h*)(pp + lo * 32 + hi * 8);
        pa.h[1] = *(const v8h*)(pp + lo * 32 + 16 + hi * 8);
        // ---- O += P * V (4 d-tiles) from LDS ----
        #pragma unroll
        for (int dt = 0; dt < 4; ++dt) {
            const _Float16* vrw = vt + (size_t)(dt * 16 + lo) * 32 + hi * 8;
            Frag16 bv;
            bv.h[0] = *(const v8h*)(vrw);
            bv.h[1] = *(const v8h*)(vrw + 16);
            o[dt] = WMMA_F16(pa.v, bv.v, o[dt]);
        }
        __syncthreads();                      // tile ib free for restaging
    }
    // ---- finalize: O /= l, store O^T into att [B,N,C] ----
    const int b = bh >> 3, h = bh & 7;
    #pragma unroll
    for (int r = 0; r < 8; ++r) {
        const float inv = 1.0f / l[r];
        const int n = n0 + r + 8 * hi;
        #pragma unroll
        for (int dt = 0; dt < 4; ++dt) {
            const int c = h * HDIM + dt * 16 + lo;
            att[((size_t)(b * NSEQ + n)) * CDIM + c] = (_Float16)(o[dt][r] * inv);
        }
    }
}

// ---------------------------------------------------------------------------
// Kernel 6: output projection.  Y[o][n] = sum_c Wo[o][c]*att[n][c] + bo[o]
// One wave -> 32 o x 64 n.  fp32 out [B,C,H,W].
// ---------------------------------------------------------------------------
__global__ __launch_bounds__(128) void k_proj_out(
    const _Float16* __restrict__ attb, const _Float16* __restrict__ Wh,
    const float* __restrict__ bias, float* __restrict__ out)
{
    const int lane = threadIdx.x & 31;
    const int wv   = threadIdx.x >> 5;
    const int waveId = blockIdx.x * 4 + wv;          // 2048 waves
    const int ngrp  = waveId & 15;
    const int otile = (waveId >> 4) & 15;
    const int b     = waveId >> 8;
    const int lo = lane & 15, hi = lane >> 4;
    const int o0 = otile * 32, n0 = ngrp * 64;

    v8f acc0[4] = {}, acc1[4] = {};
    const _Float16* arow0 = Wh + ((size_t)(o0 + lo)) * CDIM;
    const _Float16* arow1 = arow0 + (size_t)16 * CDIM;

    for (int k0 = 0; k0 < CDIM; k0 += 32) {
        Frag16 a0, a1;
        a0.h[0] = *(const v8h*)(arow0 + k0 + hi * 8);
        a0.h[1] = *(const v8h*)(arow0 + k0 + 16 + hi * 8);
        a1.h[0] = *(const v8h*)(arow1 + k0 + hi * 8);
        a1.h[1] = *(const v8h*)(arow1 + k0 + 16 + hi * 8);
        #pragma unroll
        for (int j = 0; j < 4; ++j) {
            Frag16 bf;
            const _Float16* brow = attb + ((size_t)(b * NSEQ + n0 + j * 16 + lo)) * CDIM + k0 + hi * 8;
            bf.h[0] = *(const v8h*)(brow);
            bf.h[1] = *(const v8h*)(brow + 16);
            acc0[j] = WMMA_F16(a0.v, bf.v, acc0[j]);
            acc1[j] = WMMA_F16(a1.v, bf.v, acc1[j]);
        }
    }
    #pragma unroll
    for (int r = 0; r < 8; ++r) {
        #pragma unroll
        for (int rb = 0; rb < 2; ++rb) {
            const int o = o0 + rb * 16 + r + 8 * hi;
            const float bv = bias[o];
            #pragma unroll
            for (int j = 0; j < 4; ++j) {
                const int n = n0 + j * 16 + lo;
                out[((size_t)(b * CDIM + o)) * NSEQ + n] =
                    (rb ? acc1[j][r] : acc0[j][r]) + bv;
            }
        }
    }
}

// ---------------------------------------------------------------------------
extern "C" void kernel_launch(void* const* d_in, const int* in_sizes, int n_in,
                              void* d_out, int out_size, void* d_ws, size_t ws_size,
                              hipStream_t stream)
{
    (void)in_sizes; (void)n_in; (void)out_size; (void)ws_size;
    const float* x  = (const float*)d_in[0];
    const float* Wq = (const float*)d_in[1];
    const float* bq = (const float*)d_in[2];
    const float* Wk = (const float*)d_in[3];
    const float* bk = (const float*)d_in[4];
    const float* Wv = (const float*)d_in[5];
    const float* bv = (const float*)d_in[6];
    const float* Wo = (const float*)d_in[7];
    const float* bo = (const float*)d_in[8];
    float* out = (float*)d_out;

    // workspace layout (f16 halves)
    const size_t XT  = (size_t)BATCH * NSEQ * CDIM;   // 4 Mi halves
    const size_t WSZ = (size_t)CDIM * CDIM;           // 256 Ki halves
    _Float16* xt   = (_Float16*)d_ws;
    _Float16* Wqh  = xt   + XT;
    _Float16* Wkh  = Wqh  + WSZ;
    _Float16* Wvh  = Wkh  + WSZ;
    _Float16* Woh  = Wvh  + WSZ;
    _Float16* Qb   = Woh  + WSZ;
    _Float16* Kb   = Qb   + XT;
    _Float16* Vb   = Kb   + XT;
    _Float16* attb = Vb   + XT;

    k_transpose_convert<<<dim3(NSEQ / 32, CDIM / 32, BATCH), dim3(32, 8), 0, stream>>>(x, xt);
    k_convert_f16<<<(CDIM * CDIM + 255) / 256, 256, 0, stream>>>(Wq, Wqh, CDIM * CDIM);
    k_convert_f16<<<(CDIM * CDIM + 255) / 256, 256, 0, stream>>>(Wk, Wkh, CDIM * CDIM);
    k_convert_f16<<<(CDIM * CDIM + 255) / 256, 256, 0, stream>>>(Wv, Wvh, CDIM * CDIM);
    k_convert_f16<<<(CDIM * CDIM + 255) / 256, 256, 0, stream>>>(Wo, Woh, CDIM * CDIM);

    k_proj_qk<<<512, 128, 0, stream>>>(xt, Wqh, bq, Qb, 0.125f);  // scale = d^-0.5
    k_proj_qk<<<512, 128, 0, stream>>>(xt, Wkh, bk, Kb, 1.0f);
    k_proj_v <<<512, 128, 0, stream>>>(xt, Wvh, bv, Vb);

    k_attention<<<1024, 128, 0, stream>>>(Qb, Kb, Vb, attb);

    k_proj_out<<<512, 128, 0, stream>>>(attb, Woh, bo, out);
}
